// MHESA_84885733638556
// MI455X (gfx1250) — compile-verified, matrix operand order checked by reading, not run on hardware
//
#include <hip/hip_runtime.h>

// Problem constants (from reference setup_inputs)
#define B_   4
#define N_   8192
#define D_   512
#define H_   8
#define DH_  64
#define T_   128                // scan chunk length
#define C_   (N_ / T_)          // 64 chunks
#define MTOT (B_ * N_)          // 32768 GEMM rows
#define BND  ((size_t)B_ * N_ * D_)   // 16.7M elements
#define DD   ((size_t)D_ * D_)        // 262144 weight elements

typedef __attribute__((ext_vector_type(16))) __bf16 v16bf;
typedef __attribute__((ext_vector_type(8)))  __bf16 v8bf;
typedef __attribute__((ext_vector_type(8)))  float  v8f;

// ---------------------------------------------------------------------------
// Split an f32 array into bf16 hi/lo planes: v ~= hi + lo, ~16 mantissa bits.
// ---------------------------------------------------------------------------
__global__ __launch_bounds__(256)
void split_bf16(const float* __restrict__ src, __bf16* __restrict__ hi,
                __bf16* __restrict__ lo, size_t n) {
  size_t i = (size_t)blockIdx.x * blockDim.x + threadIdx.x;
  if (i >= n) return;
  float v = src[i];
  __bf16 h = (__bf16)v;
  hi[i] = h;
  lo[i] = (__bf16)(v - (float)h);
}

// ---------------------------------------------------------------------------
// GEMM: out[M,512] = (Ahi+Alo) @ (Bhi+Blo)^T + bias with split-bf16 WMMA.
// Planes are pre-split bf16, row pitch 512. One wave = 16(M) x 64(N) tile.
// All 8 waves of a block share the same 16-row A slab (L0/L1 friendly).
// Fragment per lane (16-bit A layout): 8 contiguous K at k0+khalf and 8 more
// at k0+khalf+16, where khalf = 8*(lane>=16).  B column n == row n of W.
// K-loop fully unrolled (16 steps): all loads become base + imm offset.
// ---------------------------------------------------------------------------
__device__ __forceinline__ v16bf load_frag_bf16(const __bf16* __restrict__ p) {
  v8bf a = *(const v8bf*)(p);        // K = khalf .. khalf+7   (16B load)
  v8bf b = *(const v8bf*)(p + 16);   // K = khalf+16 .. +23    (16B load)
  v16bf f;
#pragma unroll
  for (int i = 0; i < 8; ++i) { f[i] = a[i]; f[i + 8] = b[i]; }
  return f;
}

__global__ __launch_bounds__(256)
void gemm_bt_wmma(const __bf16* __restrict__ Ahi, const __bf16* __restrict__ Alo,
                  const __bf16* __restrict__ Bhi, const __bf16* __restrict__ Blo,
                  const float* __restrict__ bias, float* __restrict__ out) {
  const int lane = threadIdx.x & 31;
  const int wave = threadIdx.x >> 5;
  const int gw   = blockIdx.x * 8 + wave;     // global wave id
  const int ntiles = D_ / 64;                 // 8 N-tiles of width 64
  const int m0 = (gw / ntiles) * 16;
  const int n0 = (gw % ntiles) * 64;

  const int lrow  = lane & 15;
  const int khalf = (lane >> 4) << 3;         // 0 or 8

  const __bf16* arow_hi = Ahi + (size_t)(m0 + lrow) * D_ + khalf;
  const __bf16* arow_lo = Alo + (size_t)(m0 + lrow) * D_ + khalf;
  const __bf16* brow_hi = Bhi + (size_t)(n0 + lrow) * D_ + khalf;
  const __bf16* brow_lo = Blo + (size_t)(n0 + lrow) * D_ + khalf;

  v8f acc[4];
#pragma unroll
  for (int ns = 0; ns < 4; ++ns)
#pragma unroll
    for (int i = 0; i < 8; ++i) acc[ns][i] = 0.0f;

#pragma unroll
  for (int k0 = 0; k0 < D_; k0 += 32) {
    v16bf a_hi = load_frag_bf16(arow_hi + k0);
    v16bf a_lo = load_frag_bf16(arow_lo + k0);
#pragma unroll
    for (int ns = 0; ns < 4; ++ns) {
      const int boff = ns * 16 * D_ + k0;     // compile-time constant
      v16bf b_hi = load_frag_bf16(brow_hi + boff);
      v16bf b_lo = load_frag_bf16(brow_lo + boff);
      // fp32 ~= hi*hi + hi*lo + lo*hi  (lo*lo ~ 2^-16, dropped)
      acc[ns] = __builtin_amdgcn_wmma_f32_16x16x32_bf16(
          false, a_hi, false, b_hi, (short)0, acc[ns], false, false);
      acc[ns] = __builtin_amdgcn_wmma_f32_16x16x32_bf16(
          false, a_hi, false, b_lo, (short)0, acc[ns], false, false);
      acc[ns] = __builtin_amdgcn_wmma_f32_16x16x32_bf16(
          false, a_lo, false, b_hi, (short)0, acc[ns], false, false);
    }
  }

  // Epilogue: D layout -> VGPR r, lane l: M = r + 8*(l>>4), N = l&15
  const int mbase = m0 + ((lane >> 4) << 3);
#pragma unroll
  for (int ns = 0; ns < 4; ++ns) {
    const int ncol = n0 + ns * 16 + lrow;
    const float bv = bias[ncol];
#pragma unroll
    for (int r = 0; r < 8; ++r)
      out[(size_t)(mbase + r) * D_ + ncol] = acc[ns][r] + bv;
  }
}

// ---------------------------------------------------------------------------
// Scan phase 1: per-chunk local scan of z[t] = (1-a) z[t-1] + a (xp[t]-xp[t-1])
// with zero start-state; emit chunk carry. Block = 64 threads (one per dh).
// ---------------------------------------------------------------------------
__global__ __launch_bounds__(64)
void scan_phase1(const float* __restrict__ xp, const float* __restrict__ init_state,
                 const float* __restrict__ alpha_param,
                 float* __restrict__ zb, float* __restrict__ carry) {
  const int id = blockIdx.x;               // [0, B*H*C)
  const int b  = id / (H_ * C_);
  const int hc = id % (H_ * C_);
  const int h  = hc / C_;
  const int ch = hc % C_;
  const int dh = threadIdx.x;

  const float a  = 1.0f / (1.0f + __expf(-alpha_param[h]));
  const float om = 1.0f - a;
  const int t0 = ch * T_;
  const size_t col = (size_t)h * DH_ + dh;
  const float* xcol = xp + (size_t)b * N_ * D_ + col;
  float* zcol       = zb + (size_t)b * N_ * D_ + col;

  float prev = (t0 == 0) ? init_state[h * DH_ + dh]
                         : xcol[(size_t)(t0 - 1) * D_];
  float z = 0.0f;
  for (int t = t0; t < t0 + T_; ++t) {
    float v = xcol[(size_t)t * D_];
    z = om * z + a * (v - prev);
    prev = v;
    zcol[(size_t)t * D_] = z;
  }
  carry[(((size_t)b * H_ + h) * C_ + ch) * DH_ + dh] = z;
}

// ---------------------------------------------------------------------------
// Scan phase 2: prefix across chunks. One thread per (b,h,dh) channel.
// s[0] = init_state; s[k+1] = (1-a)^T s[k] + carry[k].
// ---------------------------------------------------------------------------
__global__ __launch_bounds__(256)
void scan_phase2(const float* __restrict__ carry, const float* __restrict__ init_state,
                 const float* __restrict__ alpha_param, float* __restrict__ startb) {
  const int id = blockIdx.x * blockDim.x + threadIdx.x;   // [0, B*H*DH)
  if (id >= B_ * H_ * DH_) return;
  const int dh = id % DH_;
  const int bh = id / DH_;
  const int h  = bh % H_;

  const float a    = 1.0f / (1.0f + __expf(-alpha_param[h]));
  const float decT = __powf(1.0f - a, (float)T_);
  float s = init_state[h * DH_ + dh];
  const float* cr = carry  + (size_t)bh * C_ * DH_ + dh;
  float*       st = startb + (size_t)bh * C_ * DH_ + dh;
  for (int k = 0; k < C_; ++k) {
    st[(size_t)k * DH_] = s;
    s = decT * s + cr[(size_t)k * DH_];
  }
}

// ---------------------------------------------------------------------------
// Scan phase 3: z = z_local + (1-a)^{tl+1} * s_chunk, emitted directly as
// bf16 hi/lo planes (A operand of the second GEMM) — no extra pass needed.
// ---------------------------------------------------------------------------
__global__ __launch_bounds__(256)
void scan_phase3(const float* __restrict__ zb, const float* __restrict__ startb,
                 const float* __restrict__ alpha_param,
                 __bf16* __restrict__ zhi, __bf16* __restrict__ zlo) {
  const size_t e = (size_t)blockIdx.x * blockDim.x + threadIdx.x;
  if (e >= BND) return;
  const int d  = (int)(e % D_);
  const size_t bt = e / D_;
  const int t  = (int)(bt % N_);
  const int b  = (int)(bt / N_);
  const int h  = d >> 6;        // / DH_
  const int dh = d & (DH_ - 1);
  const int ch = t / T_;
  const int tl = t % T_;

  const float a = 1.0f / (1.0f + __expf(-alpha_param[h]));
  const float w = __powf(1.0f - a, (float)(tl + 1));
  const float s = startb[(((size_t)b * H_ + h) * C_ + ch) * DH_ + dh];
  const float z = zb[e] + w * s;
  __bf16 hv = (__bf16)z;
  zhi[e] = hv;
  zlo[e] = (__bf16)(z - (float)hv);
}

// ---------------------------------------------------------------------------
extern "C" void kernel_launch(void* const* d_in, const int* in_sizes, int n_in,
                              void* d_out, int out_size, void* d_ws, size_t ws_size,
                              hipStream_t stream) {
  const float* x           = (const float*)d_in[0];
  const float* W_in        = (const float*)d_in[1];
  const float* b_in        = (const float*)d_in[2];
  const float* W_out       = (const float*)d_in[3];
  const float* b_out       = (const float*)d_in[4];
  const float* init_state  = (const float*)d_in[5];
  const float* alpha_param = (const float*)d_in[6];
  float* out = (float*)d_out;

  // Workspace partition:
  //  region0 (64MB): xp (f32, GEMM1 out / scan in); reused by z_hi|z_lo planes
  //  region1 (64MB): zb (f32, local scan)
  //  region2 (64MB): x_hi | x_lo bf16 planes
  //  region3 (~3MB): carry | startb | W planes
  float*  xp    = (float*)d_ws;
  float*  zb    = xp + BND;
  __bf16* x_hi  = (__bf16*)(zb + BND);
  __bf16* x_lo  = x_hi + BND;
  float*  carry = (float*)(x_lo + BND);
  float*  startb = carry + (size_t)B_ * H_ * C_ * DH_;
  __bf16* Win_hi  = (__bf16*)(startb + (size_t)B_ * H_ * C_ * DH_);
  __bf16* Win_lo  = Win_hi + DD;
  __bf16* Wout_hi = Win_lo + DD;
  __bf16* Wout_lo = Wout_hi + DD;
  __bf16* z_hi  = (__bf16*)xp;      // reuse: xp is dead after phase1
  __bf16* z_lo  = z_hi + BND;

  const unsigned gemm_blocks = (unsigned)((MTOT / 16) * (D_ / 64) / 8);  // 2048

  // 0) one-shot bf16 hi/lo splits (weights are tiny; x streams once)
  split_bf16<<<dim3((unsigned)((DD + 255) / 256)), 256, 0, stream>>>(W_in, Win_hi, Win_lo, DD);
  split_bf16<<<dim3((unsigned)((DD + 255) / 256)), 256, 0, stream>>>(W_out, Wout_hi, Wout_lo, DD);
  split_bf16<<<dim3((unsigned)((BND + 255) / 256)), 256, 0, stream>>>(x, x_hi, x_lo, BND);

  // 1) xp = x @ W_in^T + b_in
  gemm_bt_wmma<<<gemm_blocks, 256, 0, stream>>>(x_hi, x_lo, Win_hi, Win_lo, b_in, xp);

  // 2) chunked parallel scan of the EMA-of-differences recurrence
  scan_phase1<<<dim3(B_ * H_ * C_), 64, 0, stream>>>(xp, init_state, alpha_param,
                                                     zb, carry);
  scan_phase2<<<dim3((B_ * H_ * DH_ + 255) / 256), 256, 0, stream>>>(
      carry, init_state, alpha_param, startb);
  scan_phase3<<<dim3((unsigned)((BND + 255) / 256)), 256, 0, stream>>>(
      zb, startb, alpha_param, z_hi, z_lo);

  // 3) out = z @ W_out^T + b_out
  gemm_bt_wmma<<<gemm_blocks, 256, 0, stream>>>(z_hi, z_lo, Wout_hi, Wout_lo,
                                                b_out, out);
}
